// AttentionHead_33200097198957
// MI455X (gfx1250) — matrix operand order repaired and checked by reading, not compiled, not run on hardware
//
#include <hip/hip_runtime.h>
#include <hip/hip_bf16.h>

// ---------------------------------------------------------------------------
// Attention decoder (attn-OCR style) for MI455X / gfx1250, wave32 + WMMA bf16.
// GEMM operands are pre-packed to bf16 (weights once per launch; activations
// written as bf16 by their producer kernels), so the GEMM inner loop is pure
// b128 loads + v_wmma_f32_16x16x32_bf16 with f32 accumulation.
// Each wave computes a 16x64 tile: one A fragment feeds 4 independent WMMAs.
// ---------------------------------------------------------------------------

#define Bdim 256
#define Tdim 80
#define Cdim 512
#define Hdim 512
#define NCdim 6625
#define Sdim 25
#define WIH_LD (Cdim + NCdim) // 7137

typedef __attribute__((ext_vector_type(16))) __bf16 bf16x16;
typedef __attribute__((ext_vector_type(8)))  __bf16 bf16x8;
typedef __attribute__((ext_vector_type(8)))  float  f32x8;

// ---------------------------------------------------------------------------
// GEMM: C[M,N] = A[M,K] * B[N,K]^T (+ bias[N]); A,B bf16 packed, C f32.
// One wave per 16(M) x 64(N) tile; K multiple of 32 (always 512 here).
// N tail (generator GEMM, N=6625): clamp B rows for loads, mask stores.
// All branching around the WMMAs is wave-uniform -> EXEC all ones at WMMA.
// ---------------------------------------------------------------------------
__global__ void wmma_gemm_bt(const __bf16* __restrict__ A, int lda,
                             const __bf16* __restrict__ Bm, int ldb,
                             const float* __restrict__ bias,
                             float* __restrict__ C, int ldc,
                             int M, int N, int K)
{
    const int wave = (blockIdx.x * blockDim.x + threadIdx.x) >> 5;
    const int lane = threadIdx.x & 31;
    const int nt64 = (N + 63) >> 6;             // 64-wide N tiles
    const int mt = wave / nt64;
    const int nt = wave - mt * nt64;
    if (mt * 16 >= M) return;                   // wave-uniform guard

    const int l16  = lane & 15;
    const int half = lane >> 4;

    const int arow = mt * 16 + l16;
    const __bf16* __restrict__ aptr = A + (size_t)arow * lda + 8 * half;

    // Four B row pointers (one per 16-wide sub-tile), rows clamped for loads.
    const __bf16* bptr[4];
    #pragma unroll
    for (int j = 0; j < 4; ++j) {
        int brow = nt * 64 + j * 16 + l16;
        if (brow >= N) brow = N - 1;
        bptr[j] = Bm + (size_t)brow * ldb + 8 * half;
    }

    f32x8 acc0 = {}, acc1 = {}, acc2 = {}, acc3 = {};

    for (int k = 0; k < K; k += 32) {
        if (k + 32 < K) {                       // gfx1250 global_prefetch_b8
            __builtin_prefetch(aptr + k + 32, 0, 3);
            __builtin_prefetch(bptr[0] + k + 32, 0, 3);
            __builtin_prefetch(bptr[2] + k + 32, 0, 3);
        }

        // A fragment (bf16 A 16x32 layout): lane holds K = 8h..8h+7 in elems
        // 0..7 and K = 16+8h..23+8h in elems 8..15 -> two contiguous 16B loads.
        bf16x8 alo = *(const bf16x8*)(aptr + k);
        bf16x8 ahi = *(const bf16x8*)(aptr + k + 16);
        bf16x16 va = __builtin_shufflevector(alo, ahi,
                        0, 1, 2, 3, 4, 5, 6, 7, 8, 9, 10, 11, 12, 13, 14, 15);

        // B fragments (32x16 layout): lane (n=l16, half) holds K = 8h..8h+7 and
        // K = 16+8h..23+8h of row n (same order as A) -> two 16B loads each.
        bf16x16 vb[4];
        #pragma unroll
        for (int j = 0; j < 4; ++j) {
            bf16x8 blo = *(const bf16x8*)(bptr[j] + k);
            bf16x8 bhi = *(const bf16x8*)(bptr[j] + k + 16);
            vb[j] = __builtin_shufflevector(blo, bhi,
                        0, 1, 2, 3, 4, 5, 6, 7, 8, 9, 10, 11, 12, 13, 14, 15);
        }

        // Four independent WMMAs per A fragment (no RAW between them).
        acc0 = __builtin_amdgcn_wmma_f32_16x16x32_bf16(false, va, false, vb[0], (short)0, acc0, false, false);
        acc1 = __builtin_amdgcn_wmma_f32_16x16x32_bf16(false, va, false, vb[1], (short)0, acc1, false, false);
        acc2 = __builtin_amdgcn_wmma_f32_16x16x32_bf16(false, va, false, vb[2], (short)0, acc2, false, false);
        acc3 = __builtin_amdgcn_wmma_f32_16x16x32_bf16(false, va, false, vb[3], (short)0, acc3, false, false);
    }

    // Store: elem r -> row = mt*16 + r + 8*half, col = nt*64 + j*16 + l16.
    f32x8 accs[4] = {acc0, acc1, acc2, acc3};
    #pragma unroll
    for (int j = 0; j < 4; ++j) {
        int col = nt * 64 + j * 16 + l16;
        if (col < N) {
            float bv = bias ? bias[col] : 0.0f;
            #pragma unroll
            for (int r = 0; r < 8; ++r) {
                int row = mt * 16 + r + 8 * half;
                C[(size_t)row * ldc + col] = accs[j][r] + bv;
            }
        }
    }
}

// ---------------------------------------------------------------------------
// f32 -> packed bf16 conversions (weights once per launch; inputs once).
// ---------------------------------------------------------------------------
__global__ void f32_to_bf16_kernel(const float* __restrict__ src,
                                   __bf16* __restrict__ dst, long n)
{
    long i = (long)blockIdx.x * blockDim.x + threadIdx.x;
    if (i < n) dst[i] = (__bf16)src[i];
}

// Strided variant: packs gru_wih[:, :C] (row stride 7137) into [3H, C].
__global__ void f32_to_bf16_strided_kernel(const float* __restrict__ src,
                                           __bf16* __restrict__ dst,
                                           int rows, int cols, int src_ld)
{
    long i = (long)blockIdx.x * blockDim.x + threadIdx.x;
    if (i >= (long)rows * cols) return;
    int r = (int)(i / cols), c = (int)(i - (long)r * cols);
    dst[i] = (__bf16)src[(size_t)r * src_ld + c];
}

__global__ void fill_zero_f32_kernel(float* __restrict__ p, long n)
{
    long i = (long)blockIdx.x * blockDim.x + threadIdx.x;
    if (i < n) p[i] = 0.0f;
}

__global__ void fill_zero_bf16_kernel(__bf16* __restrict__ p, long n)
{
    long i = (long)blockIdx.x * blockDim.x + threadIdx.x;
    if (i < n) p[i] = (__bf16)0.0f;
}

// ---------------------------------------------------------------------------
// Fused attention step: e[b,t] = sum_h tanh(proj[b,t,h] + prev[b,h])*sw[h];
// alpha = softmax_t(e); context[b,c] = sum_t alpha[t] * inputs[b,t,c].
// One block (256 thr = 8 waves) per batch element. Context written as bf16
// (it is only consumed as a GEMM A operand).
// ---------------------------------------------------------------------------
__global__ void attn_kernel(const float* __restrict__ proj,    // [B,T,H]
                            const float* __restrict__ prev,    // [B,H]
                            const float* __restrict__ inputs,  // [B,T,C]
                            const float* __restrict__ score_w, // [H]
                            __bf16* __restrict__ context)      // [B,C] bf16
{
    const int b   = blockIdx.x;
    const int tid = threadIdx.x;
    __shared__ float s_pp[Hdim];
    __shared__ float s_sw[Hdim];
    __shared__ float s_e[Tdim];
    __shared__ float s_alpha[Tdim];

    for (int i = tid; i < Hdim; i += 256) {
        s_pp[i] = prev[(size_t)b * Hdim + i];
        s_sw[i] = score_w[i];
    }
    __syncthreads();

    const int wave = tid >> 5, lane = tid & 31;
    const float* projb = proj + (size_t)b * Tdim * Hdim;
    for (int t = wave; t < Tdim; t += 8) {
        const float* pr = projb + (size_t)t * Hdim;
        float s = 0.0f;
        for (int h = lane; h < Hdim; h += 32)
            s += tanhf(pr[h] + s_pp[h]) * s_sw[h];
        #pragma unroll
        for (int off = 16; off > 0; off >>= 1) s += __shfl_xor(s, off, 32);
        if (lane == 0) s_e[t] = s;
    }
    __syncthreads();

    if (tid < 32) {                          // softmax over T=80 in one wave
        float m = -3.0e38f;
        for (int t = lane; t < Tdim; t += 32) m = fmaxf(m, s_e[t]);
        #pragma unroll
        for (int off = 16; off > 0; off >>= 1) m = fmaxf(m, __shfl_xor(m, off, 32));
        float sum = 0.0f;
        for (int t = lane; t < Tdim; t += 32) {
            float ev = __expf(s_e[t] - m);
            s_alpha[t] = ev;
            sum += ev;
        }
        #pragma unroll
        for (int off = 16; off > 0; off >>= 1) sum += __shfl_xor(sum, off, 32);
        float inv = 1.0f / sum;
        for (int t = lane; t < Tdim; t += 32) s_alpha[t] *= inv;
    }
    __syncthreads();

    const float* inb = inputs + (size_t)b * Tdim * Cdim;
    for (int c = tid; c < Cdim; c += 256) {
        float acc = 0.0f;
        for (int t = 0; t < Tdim; ++t)
            acc += s_alpha[t] * inb[(size_t)t * Cdim + c];
        context[(size_t)b * Cdim + c] = (__bf16)acc;
    }
}

// ---------------------------------------------------------------------------
// GRU gate fusion. The one-hot half of x@gru_wih^T is a column gather:
// gi += gru_wih[:, C + target[b]]. Biases already added by the GEMMs.
// Writes h both as f32 (recurrence/state ring) and bf16 (GEMM A operand).
// ---------------------------------------------------------------------------
__global__ void gru_fuse_kernel(const float* __restrict__ gi,      // [B,3H]
                                const float* __restrict__ gh,      // [B,3H]
                                const float* __restrict__ wih,     // [3H, WIH_LD]
                                const int*   __restrict__ targets, // [B,S]
                                const float* __restrict__ hprev,   // [B,H]
                                float* __restrict__ hnew,          // [B,H]
                                __bf16* __restrict__ hnew_bf,      // [B,H]
                                int step)
{
    int idx = blockIdx.x * blockDim.x + threadIdx.x;
    if (idx >= Bdim * Hdim) return;
    int b = idx / Hdim, j = idx - b * Hdim;
    int tgt = targets[(size_t)b * Sdim + step];
    size_t col = (size_t)Cdim + tgt;

    size_t gbase = (size_t)b * 3 * Hdim;
    float gir = gi[gbase + j]            + wih[(size_t)j * WIH_LD + col];
    float giz = gi[gbase + Hdim + j]     + wih[(size_t)(Hdim + j) * WIH_LD + col];
    float gin = gi[gbase + 2 * Hdim + j] + wih[(size_t)(2 * Hdim + j) * WIH_LD + col];
    float ghr = gh[gbase + j];
    float ghz = gh[gbase + Hdim + j];
    float ghn = gh[gbase + 2 * Hdim + j];

    float r = 1.0f / (1.0f + __expf(-(gir + ghr)));
    float z = 1.0f / (1.0f + __expf(-(giz + ghz)));
    float n = tanhf(gin + r * ghn);
    float hv = (1.0f - z) * n + z * hprev[(size_t)b * Hdim + j];
    hnew[(size_t)b * Hdim + j]    = hv;
    hnew_bf[(size_t)b * Hdim + j] = (__bf16)hv;
}

// hiddens [S,B,H] f32 (slots 1..S of ring) -> hbs [B*S, H] bf16 so the final
// generator GEMM writes d_out in [B,S,NC] order directly.
__global__ void reorder_kernel(const float* __restrict__ hidd, __bf16* __restrict__ hbs)
{
    long idx = (long)blockIdx.x * blockDim.x + threadIdx.x;
    if (idx >= (long)Sdim * Bdim * Hdim) return;
    int h = (int)(idx % Hdim);
    long t = idx / Hdim;
    int b = (int)(t % Bdim);
    int s = (int)(t / Bdim);
    hbs[((size_t)b * Sdim + s) * Hdim + h] = (__bf16)hidd[idx];
}

static void launch_gemm(const __bf16* A, int lda, const __bf16* Bm, int ldb,
                        const float* bias, float* C, int ldc,
                        int M, int N, int K, hipStream_t st)
{
    long mt = (M + 15) / 16, nt = (N + 63) / 64;
    long waves = mt * nt;
    int blocks = (int)((waves * 32 + 255) / 256);
    wmma_gemm_bt<<<blocks, 256, 0, st>>>(A, lda, Bm, ldb, bias, C, ldc, M, N, K);
}

static void launch_cvt(const float* src, __bf16* dst, long n, hipStream_t st)
{
    f32_to_bf16_kernel<<<(int)((n + 255) / 256), 256, 0, st>>>(src, dst, n);
}

extern "C" void kernel_launch(void* const* d_in, const int* in_sizes, int n_in,
                              void* d_out, int out_size, void* d_ws, size_t ws_size,
                              hipStream_t stream)
{
    const float* inputs   = (const float*)d_in[0];   // [B,T,C]
    const int*   targets  = (const int*)  d_in[1];   // [B,S]
    const float* i2h_w    = (const float*)d_in[3];   // [H,C]
    const float* h2h_w    = (const float*)d_in[4];   // [H,H]
    const float* h2h_b    = (const float*)d_in[5];   // [H]
    const float* score_w  = (const float*)d_in[6];   // [1,H]
    const float* gru_wih  = (const float*)d_in[7];   // [3H, C+NC]
    const float* gru_whh  = (const float*)d_in[8];   // [3H, H]
    const float* gru_bih  = (const float*)d_in[9];   // [3H]
    const float* gru_bhh  = (const float*)d_in[10];  // [3H]
    const float* gen_w    = (const float*)d_in[11];  // [NC,H]
    const float* gen_b    = (const float*)d_in[12];  // [NC]
    float* out = (float*)d_out;                      // [B,S,NC]

    // Workspace carve-up (256B-aligned chunks).
    char* cur = (char*)d_ws;
    auto alloc_f = [&](size_t n) { float* p = (float*)cur;  cur += (n * 4 + 255) & ~(size_t)255; return p; };
    auto alloc_b = [&](size_t n) { __bf16* p = (__bf16*)cur; cur += (n * 2 + 255) & ~(size_t)255; return p; };

    float*  proj    = alloc_f((size_t)Bdim * Tdim * Hdim);   // [B*T, H] f32
    float*  prev    = alloc_f((size_t)Bdim * Hdim);          // [B, H]   f32
    float*  gi      = alloc_f((size_t)Bdim * 3 * Hdim);      // [B, 3H]  f32
    float*  gh      = alloc_f((size_t)Bdim * 3 * Hdim);      // [B, 3H]  f32
    float*  hidd    = alloc_f((size_t)(Sdim + 1) * Bdim * Hdim); // f32 ring
    __bf16* in_bf   = alloc_b((size_t)Bdim * Tdim * Cdim);   // inputs bf16
    __bf16* i2h_bf  = alloc_b((size_t)Hdim * Cdim);
    __bf16* h2h_bf  = alloc_b((size_t)Hdim * Hdim);
    __bf16* wih_bf  = alloc_b((size_t)3 * Hdim * Cdim);      // packed [3H, C]
    __bf16* whh_bf  = alloc_b((size_t)3 * Hdim * Hdim);
    __bf16* gen_bf  = alloc_b((size_t)NCdim * Hdim);
    __bf16* ctx_bf  = alloc_b((size_t)Bdim * Cdim);
    __bf16* hbf0    = alloc_b((size_t)Bdim * Hdim);          // bf16 h ping
    __bf16* hbf1    = alloc_b((size_t)Bdim * Hdim);          // bf16 h pong
    __bf16* hbs_bf  = alloc_b((size_t)Sdim * Bdim * Hdim);   // [B*S, H] bf16

    // --- One-time packs: weights + inputs -> bf16; h0 = 0. ---
    launch_cvt(inputs, in_bf, (long)Bdim * Tdim * Cdim, stream);
    launch_cvt(i2h_w, i2h_bf, (long)Hdim * Cdim, stream);
    launch_cvt(h2h_w, h2h_bf, (long)Hdim * Hdim, stream);
    launch_cvt(gru_whh, whh_bf, (long)3 * Hdim * Hdim, stream);
    launch_cvt(gen_w, gen_bf, (long)NCdim * Hdim, stream);
    {
        long n = (long)3 * Hdim * Cdim;
        f32_to_bf16_strided_kernel<<<(int)((n + 255) / 256), 256, 0, stream>>>(
            gru_wih, wih_bf, 3 * Hdim, Cdim, WIH_LD);
    }
    fill_zero_f32_kernel<<<(Bdim * Hdim + 255) / 256, 256, 0, stream>>>(hidd, (long)Bdim * Hdim);
    fill_zero_bf16_kernel<<<(Bdim * Hdim + 255) / 256, 256, 0, stream>>>(hbf0, (long)Bdim * Hdim);

    // batch_H_proj = inputs @ i2h_w^T (no bias), hoisted out of the step loop.
    launch_gemm(in_bf, Cdim, i2h_bf, Cdim, nullptr, proj, Hdim,
                Bdim * Tdim, Hdim, Cdim, stream);

    for (int s = 0; s < Sdim; ++s) {
        const float*  h_s    = hidd + (size_t)s * Bdim * Hdim;
        float*        h_n    = hidd + (size_t)(s + 1) * Bdim * Hdim;
        const __bf16* hbf_s  = (s & 1) ? hbf1 : hbf0;
        __bf16*       hbf_n  = (s & 1) ? hbf0 : hbf1;

        // prev_proj = h @ h2h_w^T + h2h_b
        launch_gemm(hbf_s, Hdim, h2h_bf, Hdim, h2h_b, prev, Hdim,
                    Bdim, Hdim, Hdim, stream);

        // tanh-attention + softmax + context (fused); ctx emitted as bf16
        attn_kernel<<<Bdim, 256, 0, stream>>>(proj, prev, inputs, score_w, ctx_bf);

        // gi = context @ gru_wih[:, :C]^T + gru_bih  (packed bf16 weights)
        launch_gemm(ctx_bf, Cdim, wih_bf, Cdim, gru_bih, gi, 3 * Hdim,
                    Bdim, 3 * Hdim, Cdim, stream);

        // gh = h @ gru_whh^T + gru_bhh
        launch_gemm(hbf_s, Hdim, whh_bf, Hdim, gru_bhh, gh, 3 * Hdim,
                    Bdim, 3 * Hdim, Hdim, stream);

        // gates + one-hot column gather -> h_{s+1} (f32 ring + bf16 pong)
        gru_fuse_kernel<<<(Bdim * Hdim + 255) / 256, 256, 0, stream>>>(
            gi, gh, gru_wih, targets, h_s, h_n, hbf_n, s);
    }

    // [S,B,H] f32 -> [B*S,H] bf16 so the generator GEMM writes [B,S,NC].
    {
        long n = (long)Sdim * Bdim * Hdim;
        reorder_kernel<<<(int)((n + 255) / 256), 256, 0, stream>>>(
            hidd + (size_t)Bdim * Hdim, hbs_bf);
    }

    // probs = hbs @ gen_w^T + gen_b  -> d_out [B*S, NC]
    launch_gemm(hbs_bf, Hdim, gen_bf, Hdim, gen_b, out, NCdim,
                Bdim * Sdim, NCdim, Hdim, stream);
}
